// RNN_979252543773
// MI455X (gfx1250) — compile-verified, matrix operand order checked by reading, not run on hardware
//
#include <hip/hip_runtime.h>
#include <hip/hip_bf16.h>

// ---------------------------------------------------------------------------
// 2-layer tanh RNN for MI455X (gfx1250, wave32, WMMA + async-to-LDS).
//   x_t = emb[tok_t] * 32
//   h_i = tanh(h_i @ Wh_i^T + x @ Wx_i^T + b_i)    (sequential over t)
//   logits = Y @ fc_w^T + fc_b                     (one big batched GEMM,
//                                                   LDS-staged via ASYNC ops)
// ---------------------------------------------------------------------------

typedef __bf16 bf16;
typedef __attribute__((ext_vector_type(16))) __bf16 v16bf;
typedef __attribute__((ext_vector_type(8)))  float  v8f;

#define EMB_   1024
#define HID_   1024
#define SEQ_   64
#define BATCH_ 128
#define VOCAB_ 10000
#define LAYERS_ 2
#define BH_    (BATCH_ * HID_)          // 131072
#define ROWS_  (SEQ_ * BATCH_)          // 8192
#define NT_FC  625                      // 10000 / 16, exact
#define NGRP_FC 157                     // ceil(625/4)
#define ROWB   72                       // LDS tile row stride in bf16 (64 + 8 pad)

__device__ __forceinline__ bf16 f2bf(float f) {
  unsigned u = __builtin_bit_cast(unsigned, f);
  unsigned r = u + 0x7FFFu + ((u >> 16) & 1u);   // round-to-nearest-even
  unsigned short s = (unsigned short)(r >> 16);
  return __builtin_bit_cast(bf16, s);
}

union Frag16 { v16bf v; uint4 q[2]; };

// A/B fragment load per ISA 16-bit 16x32 layout:
// lane holds row (lane&15); K chunks at koff=(lane>>4)*8 : [koff..+7] and [koff+16..+23]
__device__ __forceinline__ v16bf load_frag(const bf16* p) {
  Frag16 f;
  f.q[0] = *(const uint4*)(p);
  f.q[1] = *(const uint4*)(p + 16);
  return f.v;
}

__device__ __forceinline__ v8f wmma_bf16(v16bf a, v16bf b, v8f c) {
  return __builtin_amdgcn_wmma_f32_16x16x32_bf16(
      /*neg_a=*/false, a, /*neg_b=*/false, b,
      /*c_mod=*/(short)0, c, /*reuse_a=*/false, /*reuse_b=*/false);
}

__device__ __forceinline__ void wait_async0() {
  asm volatile("s_wait_asynccnt 0" ::: "memory");
}

// ---------------------------------------------------------------------------
// fp32 -> bf16 elementwise convert
// ---------------------------------------------------------------------------
__global__ void cvt_f32_bf16_kernel(const float* __restrict__ in,
                                    bf16* __restrict__ out, int n) {
  int i = blockIdx.x * blockDim.x + threadIdx.x;
  if (i < n) out[i] = f2bf(in[i]);
}

// ---------------------------------------------------------------------------
// Embedding gather + scale by sqrt(EMB)=32, output bf16 row-major (8192 x 1024)
// ---------------------------------------------------------------------------
__global__ void embed_kernel(const int* __restrict__ toks,
                             const float* __restrict__ emb,
                             bf16* __restrict__ X) {
  int r = blockIdx.x;                       // 0..8191 == t*128 + b
  int tok = toks[r];                        // inputs is (SEQ,BATCH) row-major
  const float* src = emb + (long long)tok * EMB_;
  bf16* dst = X + (long long)r * EMB_;
  for (int c = threadIdx.x; c < EMB_; c += blockDim.x)
    dst[c] = f2bf(src[c] * 32.0f);
}

// ---------------------------------------------------------------------------
// One RNN layer step:  Hout = tanh(Hprev @ Wh^T + X @ Wx^T + bias)
// M=128 (batch), N=1024 (hid), K=1024. Each wave: 16x32 output strip.
// grid = 64 blocks x 128 threads (4 waves) -> 256 waves = 8 m-tiles x 32 n-groups.
// ---------------------------------------------------------------------------
__global__ void rnn_step_kernel(const bf16* __restrict__ Hprev,
                                const bf16* __restrict__ Wh,
                                const bf16* __restrict__ X,
                                const bf16* __restrict__ Wx,
                                const float* __restrict__ bias,
                                bf16* __restrict__ Hout,
                                bf16* __restrict__ Yout,   // may be null
                                float* __restrict__ Fout)  // may be null
{
  const int K = HID_;
  int lane = threadIdx.x & 31;
  int gw   = blockIdx.x * (blockDim.x >> 5) + (threadIdx.x >> 5);
  int mt   = gw >> 5;            // 0..7
  int ng   = gw & 31;            // 0..31, two 16-wide n-tiles each
  int lrow = lane & 15;
  int koff = (lane >> 4) << 3;

  v8f acc[2] = {};

  // Pass 1: Hprev @ Wh^T
  {
    const bf16* aB  = Hprev + (long long)(mt * 16 + lrow) * K + koff;
    const bf16* bB0 = Wh + (long long)((ng * 2 + 0) * 16 + lrow) * K + koff;
    const bf16* bB1 = Wh + (long long)((ng * 2 + 1) * 16 + lrow) * K + koff;
    for (int kt = 0; kt < K / 32; ++kt) {
      v16bf a  = load_frag(aB  + kt * 32);
      v16bf b0 = load_frag(bB0 + kt * 32);
      v16bf b1 = load_frag(bB1 + kt * 32);
      acc[0] = wmma_bf16(a, b0, acc[0]);
      acc[1] = wmma_bf16(a, b1, acc[1]);
    }
  }
  // Pass 2: X @ Wx^T (accumulate into same acc)
  {
    const bf16* aB  = X + (long long)(mt * 16 + lrow) * K + koff;
    const bf16* bB0 = Wx + (long long)((ng * 2 + 0) * 16 + lrow) * K + koff;
    const bf16* bB1 = Wx + (long long)((ng * 2 + 1) * 16 + lrow) * K + koff;
    for (int kt = 0; kt < K / 32; ++kt) {
      v16bf a  = load_frag(aB  + kt * 32);
      v16bf b0 = load_frag(bB0 + kt * 32);
      v16bf b1 = load_frag(bB1 + kt * 32);
      acc[0] = wmma_bf16(a, b0, acc[0]);
      acc[1] = wmma_bf16(a, b1, acc[1]);
    }
  }

  // Epilogue: bias + tanh; C/D layout: n = lane&15, m = v + (lane>>4)*8
  int mb = mt * 16 + ((lane >> 4) << 3);
  for (int j = 0; j < 2; ++j) {
    int n = (ng * 2 + j) * 16 + lrow;
    float bval = bias[n];
    #pragma unroll
    for (int v = 0; v < 8; ++v) {
      float val = tanhf(acc[j][v] + bval);
      int m = mb + v;
      bf16 bv = f2bf(val);
      Hout[m * HID_ + n] = bv;
      if (Yout) Yout[m * HID_ + n] = bv;
      if (Fout) Fout[m * HID_ + n] = val;
    }
  }
}

// ---------------------------------------------------------------------------
// Output projection: logits = Y @ fc_w^T + fc_b
// M=8192, N=10000 (625 n-tiles), K=1024.
// Block = 256 threads (8 waves) -> 128M x 64N tile; the 64 W rows are staged
// into double-buffered LDS with global_load_async_to_lds_b128 and shared by
// all 8 waves (8x reduction of W-side cache traffic).
// grid = 64 m-blocks x 157 n-groups = 10048 blocks.
// ---------------------------------------------------------------------------
__global__ void fc_gemm_kernel(const bf16* __restrict__ Y,
                               const bf16* __restrict__ W,
                               const float* __restrict__ bias,
                               float* __restrict__ out)
{
  __shared__ bf16 smem[2][64][ROWB];   // 2 x 64 rows x (64+8 pad) bf16 = 18 KB

  const int K = HID_;
  int tid  = threadIdx.x;
  int lane = tid & 31;
  int wv   = tid >> 5;                 // 0..7
  int mb   = blockIdx.x / NGRP_FC;     // 0..63
  int ng   = blockIdx.x % NGRP_FC;     // 0..156
  int mt   = mb * 8 + wv;              // 0..511
  int lrow = lane & 15;
  int koff = (lane >> 4) << 3;
  int nbase = ng * 64;
  int nt0   = ng * 4;
  int nvalid = NT_FC - nt0; if (nvalid > 4) nvalid = 4;

  v8f acc[4] = {};
  const bf16* aB = Y + (size_t)(mt * 16 + lrow) * K + koff;

  // Stage W tile rows [nbase, nbase+64), cols [k0, k0+64) into LDS buffer.
  // 512 chunks of 16B; 256 threads x 2 chunks. Clamp OOB rows (partial group).
  auto stage_w = [&](int k0, int buf) {
    unsigned lbase = (unsigned)(uintptr_t)(&smem[buf][0][0]);
    #pragma unroll
    for (int i = 0; i < 2; ++i) {
      int c   = tid + i * 256;         // 0..511
      int row = c >> 3;                // 0..63
      int ch  = c & 7;                 // 16B chunk within row
      int grow = nbase + row; if (grow > VOCAB_ - 1) grow = VOCAB_ - 1;
      unsigned gvoff = ((unsigned)grow * HID_ + (unsigned)(k0 + ch * 8)) * 2u;
      unsigned lds   = lbase + (unsigned)(row * (ROWB * 2) + ch * 16);
      asm volatile("global_load_async_to_lds_b128 %0, %1, %2"
                   :: "v"(lds), "v"(gvoff), "s"(W)
                   : "memory");
    }
  };

  stage_w(0, 0);
  wait_async0();
  __syncthreads();

  for (int kc = 0; kc < K / 64; ++kc) {
    int cur = kc & 1;
    if (kc + 1 < K / 64) stage_w((kc + 1) * 64, cur ^ 1);

    #pragma unroll
    for (int s = 0; s < 2; ++s) {
      int kt = kc * 2 + s;
      v16bf a = load_frag(aB + (size_t)kt * 32);
      #pragma unroll
      for (int j = 0; j < 4; ++j) {
        const bf16* p = &smem[cur][j * 16 + lrow][s * 32 + koff];
        Frag16 f;
        f.q[0] = *(const uint4*)(p);
        f.q[1] = *(const uint4*)(p + 16);
        acc[j] = wmma_bf16(a, f.v, acc[j]);
      }
    }

    if (kc + 1 < K / 64) wait_async0();
    __syncthreads();
  }

  // Epilogue: bias add + store. Wave-uniform guard (EXEC stays all-1).
  int mrow = mt * 16 + ((lane >> 4) << 3);
  for (int j = 0; j < nvalid; ++j) {
    int n = (nt0 + j) * 16 + lrow;
    float bval = bias[n];
    #pragma unroll
    for (int v = 0; v < 8; ++v) {
      int m = mrow + v;
      out[(long long)m * VOCAB_ + n] = acc[j][v] + bval;
    }
  }
}

// ---------------------------------------------------------------------------
// Host launcher
// ---------------------------------------------------------------------------
extern "C" void kernel_launch(void* const* d_in, const int* in_sizes, int n_in,
                              void* d_out, int out_size, void* d_ws, size_t ws_size,
                              hipStream_t stream) {
  const int*   toks   = (const int*)d_in[0];
  const float* hidden = (const float*)d_in[1];
  const float* emb    = (const float*)d_in[2];
  const float* Wx_w   = (const float*)d_in[3];
  const float* Wx_b   = (const float*)d_in[4];
  const float* Wh_w   = (const float*)d_in[5];
  const float* fc_w   = (const float*)d_in[6];
  const float* fc_b   = (const float*)d_in[7];
  float* out = (float*)d_out;

  // workspace carve-out (bf16 buffers)
  char* ws = (char*)d_ws;
  size_t off = 0;
  auto alloc = [&](size_t bytes) -> char* {
    char* p = ws + off;
    off += (bytes + 255) & ~(size_t)255;
    return p;
  };
  bf16* Wx_bf = (bf16*)alloc((size_t)LAYERS_ * HID_ * HID_ * sizeof(bf16));
  bf16* Wh_bf = (bf16*)alloc((size_t)LAYERS_ * HID_ * HID_ * sizeof(bf16));
  bf16* fc_bf = (bf16*)alloc((size_t)VOCAB_ * HID_ * sizeof(bf16));
  bf16* Xemb  = (bf16*)alloc((size_t)ROWS_ * EMB_ * sizeof(bf16));
  bf16* Ybf   = (bf16*)alloc((size_t)ROWS_ * HID_ * sizeof(bf16));
  bf16* hbuf  = (bf16*)alloc((size_t)2 * LAYERS_ * BH_ * sizeof(bf16)); // ping-pong
  const size_t PING = (size_t)LAYERS_ * BH_;

  const int CT = 256;
  // weight/state conversions to bf16
  int nWx = LAYERS_ * HID_ * HID_;
  cvt_f32_bf16_kernel<<<(nWx + CT - 1) / CT, CT, 0, stream>>>(Wx_w, Wx_bf, nWx);
  cvt_f32_bf16_kernel<<<(nWx + CT - 1) / CT, CT, 0, stream>>>(Wh_w, Wh_bf, nWx);
  int nFc = VOCAB_ * HID_;
  cvt_f32_bf16_kernel<<<(nFc + CT - 1) / CT, CT, 0, stream>>>(fc_w, fc_bf, nFc);
  int nH = LAYERS_ * BH_;   // initial hidden -> ping 0
  cvt_f32_bf16_kernel<<<(nH + CT - 1) / CT, CT, 0, stream>>>(hidden, hbuf, nH);

  // embedding gather + scale
  embed_kernel<<<ROWS_, 256, 0, stream>>>(toks, emb, Xemb);

  const size_t logitsN = (size_t)SEQ_ * BATCH_ * VOCAB_;
  float* hid_final = out + logitsN;   // (LAYERS, BATCH, HID)

  // sequential recurrence (stream ordering provides the dependency chain)
  for (int t = 0; t < SEQ_; ++t) {
    bf16* hprev = hbuf + (size_t)(t & 1) * PING;
    bf16* hnext = hbuf + (size_t)((t + 1) & 1) * PING;
    bool last = (t == SEQ_ - 1);

    // layer 0: x = emb row block
    rnn_step_kernel<<<64, 128, 0, stream>>>(
        hprev + 0 * BH_, Wh_bf + 0 * (size_t)HID_ * HID_,
        Xemb + (size_t)t * BH_, Wx_bf + 0 * (size_t)HID_ * HID_,
        Wx_b + 0 * HID_,
        hnext + 0 * BH_, (bf16*)nullptr,
        last ? (hid_final + 0 * BH_) : (float*)nullptr);

    // layer 1: x = layer-0 output; also emit bf16 row block into Y for fc GEMM
    rnn_step_kernel<<<64, 128, 0, stream>>>(
        hprev + 1 * BH_, Wh_bf + 1 * (size_t)HID_ * HID_,
        hnext + 0 * BH_, Wx_bf + 1 * (size_t)HID_ * HID_,
        Wx_b + 1 * HID_,
        hnext + 1 * BH_, Ybf + (size_t)t * BH_,
        last ? (hid_final + 1 * BH_) : (float*)nullptr);
  }

  // batched output projection over all timesteps: (8192 x 1024) @ (1024 x 10000)
  fc_gemm_kernel<<<64 * NGRP_FC, 256, 0, stream>>>(Ybf, fc_bf, fc_b, out);
}
